// Block_67577015435861
// MI455X (gfx1250) — matrix operand order repaired and checked by reading, not compiled, not run on hardware
//
#include <hip/hip_runtime.h>

// ---------------------------------------------------------------------------
// Transformer block (MHSA + FFN + 2x LayerNorm) for MI455X / gfx1250.
// All matmuls on v_wmma_f32_16x16x32_bf16 (fp32 accumulate).  Weights are
// pre-packed into WMMA fragment-major bf16 so every B fragment is two aligned
// b128 loads; A operands are bf16 row-major loaded as two b128 loads.
// GEMM waves own 32x64 output tiles: 12 b128 loads per 8 WMMAs.
// ---------------------------------------------------------------------------

#define D_MODEL 512
#define DFF     2048
#define NH      8
#define D_HEAD  64
#define TT      256
#define BB      64
#define MM      (BB * TT)   // 16384 rows

typedef __bf16 bf16_t;
typedef __attribute__((ext_vector_type(16))) __bf16 v16bf;
typedef __attribute__((ext_vector_type(8)))  __bf16 v8bf;
typedef __attribute__((ext_vector_type(8)))  float  v8f;

__device__ __forceinline__ v8f wmma_bf16(v16bf a, v16bf b, v8f c) {
  return __builtin_amdgcn_wmma_f32_16x16x32_bf16(false, a, false, b,
                                                 (short)0, c, false, false);
}

__device__ __forceinline__ v16bf cat8(v8bf lo, v8bf hi) {
  return __builtin_shufflevector(lo, hi, 0, 1, 2, 3, 4, 5, 6, 7,
                                 8, 9, 10, 11, 12, 13, 14, 15);
}

// ---- fragment loaders (contiguous k = 16*(lane/16) + s), all 2x 16-byte vector loads ----
__device__ __forceinline__ v16bf load_a(const bf16_t* A, int lda, int row0,
                                        int k0, int lane) {
  const bf16_t* p = A + (size_t)(row0 + (lane & 15)) * lda + k0 + ((lane >> 4) << 4);
  return cat8(*(const v8bf*)p, *(const v8bf*)(p + 8));
}
// B[k][n] = S[(n0+n)*ld + k0 + k] (16 contiguous bf16 per lane).
__device__ __forceinline__ v16bf load_bT(const bf16_t* S, int ld, int n0,
                                         int k0, int lane) {
  const bf16_t* p = S + (size_t)(n0 + (lane & 15)) * ld + k0 + ((lane >> 4) << 4);
  return cat8(*(const v8bf*)p, *(const v8bf*)(p + 8));
}
// Packed fragment-major B: tile (tk, tn) stored as 32 lanes x 16 bf16.
__device__ __forceinline__ v16bf load_bp(const bf16_t* P, int tn_count, int tk,
                                         int tn, int lane) {
  const bf16_t* p = P + (((size_t)tk * tn_count + tn) * 32 + lane) * 16;
  return cat8(*(const v8bf*)p, *(const v8bf*)(p + 8));
}

// ---------------------------------------------------------------------------
// Pack kernel: fp32 weight [K,N] (Z matrices) -> fragment-major bf16.
// grid (K/32, N/16, Z), block 32.  One-time gather cost.
// ---------------------------------------------------------------------------
__global__ void k_pack_b(const float* __restrict__ src, bf16_t* __restrict__ dst,
                         int K, int N, size_t sz, size_t dz) {
  int lane = threadIdx.x & 31;
  const float* S = src + (size_t)blockIdx.z * sz;
  bf16_t* D = dst + (size_t)blockIdx.z * dz;
  int tk = blockIdx.x, tn = blockIdx.y;
  int col = tn * 16 + (lane & 15);
  int kb = tk * 32 + ((lane >> 4) << 4);
  bf16_t* d = D + (((size_t)tk * (N >> 4) + tn) * 32 + lane) * 16;
#pragma unroll
  for (int i = 0; i < 16; ++i) d[i] = (bf16_t)S[(size_t)(kb + i) * N + col];
}

// fp32 -> bf16 elementwise. grid (n/1024), block 256, 4/thread.
__global__ void k_f2bf(const float* __restrict__ src, bf16_t* __restrict__ dst) {
  size_t i = ((size_t)blockIdx.x * 256 + threadIdx.x) * 4;
#pragma unroll
  for (int j = 0; j < 4; ++j) dst[i + j] = (bf16_t)src[i + j];
}

// ---------------------------------------------------------------------------
// Kernel 1: per-head Q/K/V projection, 32 rows per wave.
// q/k row-major [B,H,T,64]; V transposed [B,H,64,256].  grid (M/32, H, 3).
// ---------------------------------------------------------------------------
__global__ void k_qkv(const bf16_t* __restrict__ xb, const bf16_t* __restrict__ WqP,
                      const bf16_t* __restrict__ WkP, const bf16_t* __restrict__ WvP,
                      bf16_t* __restrict__ q, bf16_t* __restrict__ kq,
                      bf16_t* __restrict__ vt) {
  int lane = threadIdx.x & 31;
  int h = blockIdx.y, z = blockIdx.z;
  const bf16_t* WP = (z == 0 ? WqP : z == 1 ? WkP : WvP) + (size_t)h * D_MODEL * D_HEAD;
  int row0 = blockIdx.x * 32;

  v8f acc0[4] = {}, acc1[4] = {};
  for (int k0 = 0; k0 < D_MODEL; k0 += 32) {
    v16bf a0 = load_a(xb, D_MODEL, row0, k0, lane);
    v16bf a1 = load_a(xb, D_MODEL, row0 + 16, k0, lane);
#pragma unroll
    for (int j = 0; j < 4; ++j) {
      v16bf b = load_bp(WP, D_HEAD / 16, k0 >> 5, j, lane);
      acc0[j] = wmma_bf16(a0, b, acc0[j]);
      acc1[j] = wmma_bf16(a1, b, acc1[j]);
    }
  }
  int n_ = lane & 15, mb = (lane >> 4) << 3;
  int bidx = row0 >> 8, t0 = row0 & 255;            // row = b*256 + t
  if (z < 2) {
    bf16_t* ob = (z == 0 ? q : kq) + (((size_t)bidx * NH + h) * TT + t0) * D_HEAD;
#pragma unroll
    for (int j = 0; j < 4; ++j)
#pragma unroll
      for (int r = 0; r < 8; ++r) {
        ob[(size_t)(mb + r) * D_HEAD + j * 16 + n_]      = (bf16_t)acc0[j][r];
        ob[(size_t)(16 + mb + r) * D_HEAD + j * 16 + n_] = (bf16_t)acc1[j][r];
      }
  } else {                                          // V transposed: [.,64,256]
    bf16_t* ob = vt + (((size_t)bidx * NH + h) * D_HEAD) * TT;
#pragma unroll
    for (int j = 0; j < 4; ++j)
#pragma unroll
      for (int r = 0; r < 8; ++r) {
        ob[(size_t)(j * 16 + n_) * TT + t0 + mb + r]      = (bf16_t)acc0[j][r];
        ob[(size_t)(j * 16 + n_) * TT + t0 + 16 + mb + r] = (bf16_t)acc1[j][r];
      }
  }
}

// ---------------------------------------------------------------------------
// Kernel 2: attention for one (b, h, 16-query tile).  grid (T/16, H, B),
// block 32.  Scores -> causal softmax in LDS -> P @ V (V transposed).
// ---------------------------------------------------------------------------
__global__ void k_attn(const bf16_t* __restrict__ q, const bf16_t* __restrict__ kq,
                       const bf16_t* __restrict__ vt, bf16_t* __restrict__ heads) {
  __shared__ float  sc[16][TT];   // 16 KB
  __shared__ bf16_t pr[16][TT];   // 8 KB
  int lane = threadIdx.x & 31;
  int qt = blockIdx.x, h = blockIdx.y, b = blockIdx.z;
  size_t base = ((size_t)b * NH + h) * TT * D_HEAD;
  const bf16_t* Qb = q + base;
  const bf16_t* Kb = kq + base;
  const bf16_t* Vt = vt + base;   // [64][256]
  int row0 = qt * 16;
  int n_ = lane & 15, mb = (lane >> 4) << 3;

  v16bf aq0 = load_a(Qb, D_HEAD, row0, 0, lane);
  v16bf aq1 = load_a(Qb, D_HEAD, row0, 32, lane);

  for (int kt = 0; kt < TT / 16; ++kt) {
    v8f acc = {};
    acc = wmma_bf16(aq0, load_bT(Kb, D_HEAD, kt * 16, 0, lane), acc);
    acc = wmma_bf16(aq1, load_bT(Kb, D_HEAD, kt * 16, 32, lane), acc);
    int key = kt * 16 + n_;
#pragma unroll
    for (int r = 0; r < 8; ++r) {
      int m = mb + r;
      float s = acc[r] * 0.125f;                       // D_HEAD^-0.5
      sc[m][key] = (key <= row0 + m) ? s : -1.0e30f;   // causal mask
    }
  }
  __syncthreads();

  if (lane < 16) {                                     // softmax, one row/lane
    int m = lane;
    float mx = -1.0e30f;
    for (int c = 0; c < TT; ++c) mx = fmaxf(mx, sc[m][c]);
    float sum = 0.f;
    for (int c = 0; c < TT; ++c) { float e = __expf(sc[m][c] - mx); sc[m][c] = e; sum += e; }
    float rinv = 1.0f / sum;
    for (int c = 0; c < TT; ++c) pr[m][c] = (bf16_t)(sc[m][c] * rinv);
  }
  __syncthreads();

  v8f av[4] = {};
  for (int kc = 0; kc < TT / 32; ++kc) {
    v16bf ap = load_a(&pr[0][0], TT, 0, kc * 32, lane);   // ds_load_b128 x2
#pragma unroll
    for (int j = 0; j < 4; ++j)
      av[j] = wmma_bf16(ap, load_bT(Vt, TT, j * 16, kc * 32, lane), av[j]);
  }
  bf16_t* ob = heads + ((size_t)b * TT + row0) * (NH * D_HEAD) + h * D_HEAD;
#pragma unroll
  for (int j = 0; j < 4; ++j)
#pragma unroll
    for (int r = 0; r < 8; ++r)
      ob[(size_t)(mb + r) * (NH * D_HEAD) + j * 16 + n_] = (bf16_t)av[j][r];
}

__device__ __forceinline__ void wave_sum2(float& s, float& s2) {
#pragma unroll
  for (int o = 16; o >= 1; o >>= 1) {
    s  += __shfl_xor(s,  o, 32);
    s2 += __shfl_xor(s2, o, 32);
  }
}

// ---------------------------------------------------------------------------
// Kernel 3: heads @ Wo + x residual -> LayerNorm1 (fp32 + bf16 copies).
// grid (M/32), block 256 (8 waves x 64 cols), 32 rows per block.
// ---------------------------------------------------------------------------
__global__ void k_oproj_ln(const bf16_t* __restrict__ heads, const bf16_t* __restrict__ WoP,
                           const float* __restrict__ x, const float* __restrict__ g,
                           const float* __restrict__ beta, float* __restrict__ ln1,
                           bf16_t* __restrict__ ln1b) {
  __shared__ float buf[32][D_MODEL];   // 64 KB
  int lane = threadIdx.x & 31, wave = threadIdx.x >> 5;
  int row0 = blockIdx.x * 32;
  int col0 = wave * 64;

  v8f acc0[4] = {}, acc1[4] = {};
  for (int k0 = 0; k0 < D_MODEL; k0 += 32) {
    v16bf a0 = load_a(heads, D_MODEL, row0, k0, lane);
    v16bf a1 = load_a(heads, D_MODEL, row0 + 16, k0, lane);
#pragma unroll
    for (int j = 0; j < 4; ++j) {
      v16bf b = load_bp(WoP, D_MODEL / 16, k0 >> 5, (col0 >> 4) + j, lane);
      acc0[j] = wmma_bf16(a0, b, acc0[j]);
      acc1[j] = wmma_bf16(a1, b, acc1[j]);
    }
  }
  int n_ = lane & 15, mb = (lane >> 4) << 3;
#pragma unroll
  for (int j = 0; j < 4; ++j)
#pragma unroll
    for (int r = 0; r < 8; ++r) {
      int m = mb + r, c = col0 + j * 16 + n_;
      buf[m][c]      = acc0[j][r] + x[(size_t)(row0 + m) * D_MODEL + c];
      buf[m + 16][c] = acc1[j][r] + x[(size_t)(row0 + 16 + m) * D_MODEL + c];
    }
  __syncthreads();

#pragma unroll
  for (int rr = 0; rr < 4; ++rr) {
    int m = wave * 4 + rr;
    float s = 0.f, s2 = 0.f;
#pragma unroll
    for (int i = 0; i < 16; ++i) {
      float vv = buf[m][lane + 32 * i];
      s += vv; s2 += vv * vv;
    }
    wave_sum2(s, s2);
    float mean = s * (1.f / D_MODEL);
    float var  = s2 * (1.f / D_MODEL) - mean * mean;
    float rstd = rsqrtf(var + 1e-5f);
#pragma unroll
    for (int i = 0; i < 16; ++i) {
      int c = lane + 32 * i;
      float o = (buf[m][c] - mean) * rstd * g[c] + beta[c];
      ln1[(size_t)(row0 + m) * D_MODEL + c]  = o;
      ln1b[(size_t)(row0 + m) * D_MODEL + c] = (bf16_t)o;
    }
  }
}

// ---------------------------------------------------------------------------
// Kernel 4: h = relu(ln1 @ W1 + b1), bf16 out.  grid (M/32, DFF/512), block 256.
// ---------------------------------------------------------------------------
__global__ void k_ffn1(const bf16_t* __restrict__ ln1b, const bf16_t* __restrict__ W1P,
                       const float* __restrict__ b1, bf16_t* __restrict__ hbuf) {
  int lane = threadIdx.x & 31, wave = threadIdx.x >> 5;
  int row0 = blockIdx.x * 32;
  int col0 = blockIdx.y * 512 + wave * 64;

  v8f acc0[4] = {}, acc1[4] = {};
  for (int k0 = 0; k0 < D_MODEL; k0 += 32) {
    v16bf a0 = load_a(ln1b, D_MODEL, row0, k0, lane);
    v16bf a1 = load_a(ln1b, D_MODEL, row0 + 16, k0, lane);
#pragma unroll
    for (int j = 0; j < 4; ++j) {
      v16bf b = load_bp(W1P, DFF / 16, k0 >> 5, (col0 >> 4) + j, lane);
      acc0[j] = wmma_bf16(a0, b, acc0[j]);
      acc1[j] = wmma_bf16(a1, b, acc1[j]);
    }
  }
  int n_ = lane & 15, mb = (lane >> 4) << 3;
#pragma unroll
  for (int j = 0; j < 4; ++j)
#pragma unroll
    for (int r = 0; r < 8; ++r) {
      int c = col0 + j * 16 + n_;
      float bias = b1[c];
      hbuf[(size_t)(row0 + mb + r) * DFF + c] =
          (bf16_t)fmaxf(acc0[j][r] + bias, 0.f);
      hbuf[(size_t)(row0 + 16 + mb + r) * DFF + c] =
          (bf16_t)fmaxf(acc1[j][r] + bias, 0.f);
    }
}

// ---------------------------------------------------------------------------
// Kernel 5: h @ W2 + b2 + ln1 residual -> LayerNorm2 -> d_out.  grid (M/32),
// block 256, K = 2048.
// ---------------------------------------------------------------------------
__global__ void k_ffn2_ln(const bf16_t* __restrict__ hbuf, const bf16_t* __restrict__ W2P,
                          const float* __restrict__ b2, const float* __restrict__ ln1,
                          const float* __restrict__ g, const float* __restrict__ beta,
                          float* __restrict__ out) {
  __shared__ float buf[32][D_MODEL];   // 64 KB
  int lane = threadIdx.x & 31, wave = threadIdx.x >> 5;
  int row0 = blockIdx.x * 32;
  int col0 = wave * 64;

  v8f acc0[4] = {}, acc1[4] = {};
  for (int k0 = 0; k0 < DFF; k0 += 32) {
    v16bf a0 = load_a(hbuf, DFF, row0, k0, lane);
    v16bf a1 = load_a(hbuf, DFF, row0 + 16, k0, lane);
#pragma unroll
    for (int j = 0; j < 4; ++j) {
      v16bf b = load_bp(W2P, D_MODEL / 16, k0 >> 5, (col0 >> 4) + j, lane);
      acc0[j] = wmma_bf16(a0, b, acc0[j]);
      acc1[j] = wmma_bf16(a1, b, acc1[j]);
    }
  }
  int n_ = lane & 15, mb = (lane >> 4) << 3;
#pragma unroll
  for (int j = 0; j < 4; ++j)
#pragma unroll
    for (int r = 0; r < 8; ++r) {
      int m = mb + r, c = col0 + j * 16 + n_;
      buf[m][c]      = acc0[j][r] + b2[c] + ln1[(size_t)(row0 + m) * D_MODEL + c];
      buf[m + 16][c] = acc1[j][r] + b2[c] + ln1[(size_t)(row0 + 16 + m) * D_MODEL + c];
    }
  __syncthreads();

#pragma unroll
  for (int rr = 0; rr < 4; ++rr) {
    int m = wave * 4 + rr;
    float s = 0.f, s2 = 0.f;
#pragma unroll
    for (int i = 0; i < 16; ++i) {
      float vv = buf[m][lane + 32 * i];
      s += vv; s2 += vv * vv;
    }
    wave_sum2(s, s2);
    float mean = s * (1.f / D_MODEL);
    float var  = s2 * (1.f / D_MODEL) - mean * mean;
    float rstd = rsqrtf(var + 1e-5f);
#pragma unroll
    for (int i = 0; i < 16; ++i) {
      int c = lane + 32 * i;
      out[(size_t)(row0 + m) * D_MODEL + c] = (buf[m][c] - mean) * rstd * g[c] + beta[c];
    }
  }
}

// ---------------------------------------------------------------------------
extern "C" void kernel_launch(void* const* d_in, const int* in_sizes, int n_in,
                              void* d_out, int out_size, void* d_ws, size_t ws_size,
                              hipStream_t stream) {
  (void)in_sizes; (void)n_in; (void)out_size; (void)ws_size;
  const float* x    = (const float*)d_in[0];
  const float* Wq   = (const float*)d_in[1];
  const float* Wk   = (const float*)d_in[2];
  const float* Wv   = (const float*)d_in[3];
  const float* Wo   = (const float*)d_in[4];
  const float* W1   = (const float*)d_in[5];
  const float* b1   = (const float*)d_in[6];
  const float* W2   = (const float*)d_in[7];
  const float* b2   = (const float*)d_in[8];
  const float* ln1g = (const float*)d_in[9];
  const float* ln1b_= (const float*)d_in[10];
  const float* ln2g = (const float*)d_in[11];
  const float* ln2b = (const float*)d_in[12];

  size_t off = 0;
  char* base = (char*)d_ws;
  auto carve = [&](size_t bytes) -> void* {
    void* p = base + off;
    off += (bytes + 255) & ~(size_t)255;
    return p;
  };
  const size_t nQKV  = (size_t)BB * NH * TT * D_HEAD;       // 8.39M
  const size_t nROWS = (size_t)MM * D_MODEL;                // 8.39M
  const size_t nHW   = (size_t)NH * D_MODEL * D_HEAD;
  bf16_t* xb    = (bf16_t*)carve(nROWS * sizeof(bf16_t));
  bf16_t* WqP   = (bf16_t*)carve(nHW * sizeof(bf16_t));
  bf16_t* WkP   = (bf16_t*)carve(nHW * sizeof(bf16_t));
  bf16_t* WvP   = (bf16_t*)carve(nHW * sizeof(bf16_t));
  bf16_t* WoP   = (bf16_t*)carve((size_t)D_MODEL * D_MODEL * sizeof(bf16_t));
  bf16_t* W1P   = (bf16_t*)carve((size_t)D_MODEL * DFF * sizeof(bf16_t));
  bf16_t* W2P   = (bf16_t*)carve((size_t)DFF * D_MODEL * sizeof(bf16_t));
  bf16_t* q     = (bf16_t*)carve(nQKV * sizeof(bf16_t));
  bf16_t* kq    = (bf16_t*)carve(nQKV * sizeof(bf16_t));
  bf16_t* vt    = (bf16_t*)carve(nQKV * sizeof(bf16_t));
  bf16_t* heads = (bf16_t*)carve(nROWS * sizeof(bf16_t));
  float*  ln1   = (float*)carve(nROWS * sizeof(float));
  bf16_t* ln1bf = (bf16_t*)carve(nROWS * sizeof(bf16_t));
  bf16_t* hbuf  = (bf16_t*)carve((size_t)MM * DFF * sizeof(bf16_t));

  // --- one-time packing / conversion ---
  size_t whz = (size_t)D_MODEL * D_HEAD;
  k_f2bf  <<<dim3(nROWS / 1024), 256, 0, stream>>>(x, xb);
  k_pack_b<<<dim3(16, 4, NH),  32, 0, stream>>>(Wq, WqP, D_MODEL, D_HEAD, whz, whz);
  k_pack_b<<<dim3(16, 4, NH),  32, 0, stream>>>(Wk, WkP, D_MODEL, D_HEAD, whz, whz);
  k_pack_b<<<dim3(16, 4, NH),  32, 0, stream>>>(Wv, WvP, D_MODEL, D_HEAD, whz, whz);
  k_pack_b<<<dim3(16, 32, 1),  32, 0, stream>>>(Wo, WoP, D_MODEL, D_MODEL, 0, 0);
  k_pack_b<<<dim3(16, 128, 1), 32, 0, stream>>>(W1, W1P, D_MODEL, DFF, 0, 0);
  k_pack_b<<<dim3(64, 32, 1),  32, 0, stream>>>(W2, W2P, DFF, D_MODEL, 0, 0);

  // --- main pipeline ---
  k_qkv     <<<dim3(MM / 32, NH, 3), 32,  0, stream>>>(xb, WqP, WkP, WvP, q, kq, vt);
  k_attn    <<<dim3(TT / 16, NH, BB), 32, 0, stream>>>(q, kq, vt, heads);
  k_oproj_ln<<<dim3(MM / 32), 256,        0, stream>>>(heads, WoP, x, ln1g, ln1b_, ln1, ln1bf);
  k_ffn1    <<<dim3(MM / 32, DFF / 512), 256, 0, stream>>>(ln1bf, W1P, b1, hbuf);
  k_ffn2_ln <<<dim3(MM / 32), 256,        0, stream>>>(hbuf, W2P, b2, ln1, ln2g, ln2b,
                                                       (float*)d_out);
}
